// CLIPWrapper_77549929496840
// MI455X (gfx1250) — matrix operand order repaired and checked by reading, not compiled, not run on hardware
//
#include <hip/hip_runtime.h>
#include <hip/hip_bf16.h>
#include <cstdint>

// ---------------------------------------------------------------------------
// CLIP ViT-B/16 visual tower forward on gfx1250 (MI455X).
// Matmuls: v_wmma_f32_16x16x32_bf16, 32x32 wave tiles, double-buffered LDS
// staging fed by GLOBAL_LOAD_ASYNC_TO_LDS_B128 (ASYNCcnt-tracked DMA).
// ---------------------------------------------------------------------------

typedef __attribute__((ext_vector_type(16))) __bf16 v16bf;
typedef __attribute__((ext_vector_type(8)))  __bf16 v8bf;
typedef __attribute__((ext_vector_type(8)))  float  v8f;

#define BB   32      // batch
#define SS   197     // valid tokens
#define SP   256     // padded tokens (multiple of 64)
#define DD   768     // width
#define HH   12      // heads
#define HDIM 64      // head dim
#define FF   3072    // mlp dim
#define LL   12      // layers
#define TD   2304    // 3*D
#define MP   (BB*SP) // 8192 padded token rows
#define PV   (BB*196)// 6272 valid patch rows
#define PM   6336    // patch rows padded to mult of 64
#define BH   (BB*HH) // 384 attention batches
#define OO   512     // output proj dim

// ---- CDNA5 async global->LDS 16B copy (tracked with ASYNCcnt) -------------
__device__ __forceinline__ void async_ld16(void* lds, const void* g)
{
    unsigned l = (unsigned)(uintptr_t)lds;          // addr[31:0] = LDS offset
    unsigned long long ga = (unsigned long long)(uintptr_t)g;
    asm volatile("global_load_async_to_lds_b128 %0, %1, off"
                 :: "v"(l), "v"(ga) : "memory");
}
__device__ __forceinline__ void wait_async0()
{
    asm volatile("s_wait_asynccnt 0" ::: "memory");
}

// ---------------------------------------------------------------------------
// Generic batched GEMM: C[m,n] = act( A[m,:K] . Bw[n,:K] + bias[n] + res[m,n] )
// A: bf16 row-major [M,K]; Bw: bf16 row-major [N,K] (i.e. computes A @ Bw^T).
// Block = 256 thr = 8 waves (2 M-waves x 4 N-waves); block tile 64 x 128;
// wave tile 32x32 = 4 accumulators; K-step 32; double-buffered async LDS.
// ---------------------------------------------------------------------------
__global__ __launch_bounds__(256) void gemm_bf16_kernel(
    const __bf16* __restrict__ A, long sA,
    const __bf16* __restrict__ Bw, long sB,
    float* __restrict__ outF, __bf16* __restrict__ outB, long sC,
    const float* __restrict__ bias,
    const float* __restrict__ res, long sR,
    int M, int N, int K, int Mvalid, int act)
{
    __shared__ __attribute__((aligned(16))) __bf16 Asb[2][64 * 32];
    __shared__ __attribute__((aligned(16))) __bf16 Bsb[2][128 * 32];

    const int bz = blockIdx.z;
    A  += (long)bz * sA;
    Bw += (long)bz * sB;
    const long cbase = (long)bz * sC;

    const int m0   = blockIdx.y * 64;
    const int n0   = blockIdx.x * 128;
    const int tid  = threadIdx.x;
    const int lane = tid & 31;
    const int w    = tid >> 5;
    const int wm   = w & 1;        // 0..1 -> 32-row slab
    const int wn   = w >> 1;       // 0..3 -> 32-col slab

    // A-tile staging: 64x32 bf16 = 4KB; one 16B async copy per thread.
    const int arow = tid >> 2;
    const int acol = (tid & 3) << 3;
    const long ga_off = (long)(m0 + arow) * K + acol;

    // Fragment addressing (16-bit A layout / B layout, wave32).
    const int lcol = (lane >> 4) << 3;        // A: K 0-7 | 8-15 half select
    const int bko  = (lane >> 4) << 4;        // unused: B frags use lcol, same k order as A
    const int lr0  = wm * 32 + (lane & 15);   // A frag row (mi=0)
    const int bc0  = wn * 32 + (lane & 15);   // B frag col, tile-local (ni=0)

    auto stage = [&](int buf, int k0) {
        async_ld16(&Asb[buf][arow * 32 + acol], A + ga_off + k0);
#pragma unroll
        for (int j = 0; j < 2; ++j) {          // B tile 128x32 = 8KB
            int c   = tid + (j << 8);
            int col = c >> 2;
            int off = (c & 3) << 3;
            int gcol = n0 + col; if (gcol > N - 1) gcol = N - 1;
            async_ld16(&Bsb[buf][col * 32 + off],
                       Bw + (long)gcol * K + k0 + off);
        }
    };

    v8f acc[2][2] = {};
    const int nst = K >> 5;

    stage(0, 0);
    wait_async0();
    __syncthreads();

    for (int s = 0; s < nst; ++s) {
        const int buf = s & 1;
        const int k0  = s << 5;
        if (s + 1 < nst) stage(buf ^ 1, k0 + 32);   // DMA overlaps compute
        if (k0 + 96 <= K) {                          // GL2 prefetch, +2 stages
            int col = tid >> 2, off = (tid & 3) << 3;
            int gcol = n0 + col; if (gcol > N - 1) gcol = N - 1;
            __builtin_prefetch(Bw + (long)gcol * K + k0 + 64 + off, 0, 1);
        }

        v16bf a0, a1, b0, b1;
        ((v8bf*)&a0)[0] = *(const v8bf*)&Asb[buf][lr0 * 32 + lcol];
        ((v8bf*)&a0)[1] = *(const v8bf*)&Asb[buf][lr0 * 32 + lcol + 16];
        ((v8bf*)&a1)[0] = *(const v8bf*)&Asb[buf][(lr0 + 16) * 32 + lcol];
        ((v8bf*)&a1)[1] = *(const v8bf*)&Asb[buf][(lr0 + 16) * 32 + lcol + 16];
        ((v8bf*)&b0)[0] = *(const v8bf*)&Bsb[buf][bc0 * 32 + lcol];
        ((v8bf*)&b0)[1] = *(const v8bf*)&Bsb[buf][bc0 * 32 + lcol + 16];
        ((v8bf*)&b1)[0] = *(const v8bf*)&Bsb[buf][(bc0 + 16) * 32 + lcol];
        ((v8bf*)&b1)[1] = *(const v8bf*)&Bsb[buf][(bc0 + 16) * 32 + lcol + 16];

        acc[0][0] = __builtin_amdgcn_wmma_f32_16x16x32_bf16(
                        false, a0, false, b0, (short)0, acc[0][0], false, false);
        acc[0][1] = __builtin_amdgcn_wmma_f32_16x16x32_bf16(
                        false, a0, false, b1, (short)0, acc[0][1], false, false);
        acc[1][0] = __builtin_amdgcn_wmma_f32_16x16x32_bf16(
                        false, a1, false, b0, (short)0, acc[1][0], false, false);
        acc[1][1] = __builtin_amdgcn_wmma_f32_16x16x32_bf16(
                        false, a1, false, b1, (short)0, acc[1][1], false, false);

        wait_async0();        // next-stage tiles landed
        __syncthreads();      // visible to all waves; current tiles consumed
    }

    // C/D layout: VGPR r, lanes0-15 -> (M=r, N=lane); lanes16-31 -> (M=r+8).
#pragma unroll
    for (int mi = 0; mi < 2; ++mi) {
#pragma unroll
        for (int ni = 0; ni < 2; ++ni) {
            const int rb = m0 + wm * 32 + mi * 16 + ((lane >> 4) << 3);
            const int cc = n0 + wn * 32 + ni * 16 + (lane & 15);
            if (cc >= N) continue;
#pragma unroll
            for (int r = 0; r < 8; ++r) {
                int row = rb + r;
                if (row >= Mvalid) continue;
                float v = acc[mi][ni][r];
                if (bias) v += bias[cc];
                if (res)  v += res[(long)bz * sR + (long)row * N + cc];
                if (act == 1)  // QuickGELU
                    v = v / (1.f + __expf(-1.702f * v));
                long o = cbase + (long)row * N + cc;
                if (outF) outF[o] = v;
                if (outB) outB[o] = (__bf16)v;
            }
        }
    }
}

// ---------------------------------------------------------------------------
// LayerNorm over D per row; optional fp32 and/or bf16 output.
// ---------------------------------------------------------------------------
__global__ __launch_bounds__(256) void layernorm_kernel(
    const float* __restrict__ x, const float* __restrict__ gamma,
    const float* __restrict__ beta, float* __restrict__ outF,
    __bf16* __restrict__ outB, int D)
{
    const long row = blockIdx.x;
    const float* xr = x + row * D;
    __shared__ float red[256];
    const int tid = threadIdx.x;

    float s = 0.f;
    for (int i = tid; i < D; i += 256) s += xr[i];
    red[tid] = s; __syncthreads();
    for (int o = 128; o > 0; o >>= 1) {
        if (tid < o) red[tid] += red[tid + o];
        __syncthreads();
    }
    const float mean = red[0] / (float)D;
    __syncthreads();

    float v = 0.f;
    for (int i = tid; i < D; i += 256) { float d = xr[i] - mean; v += d * d; }
    red[tid] = v; __syncthreads();
    for (int o = 128; o > 0; o >>= 1) {
        if (tid < o) red[tid] += red[tid + o];
        __syncthreads();
    }
    const float rstd = rsqrtf(red[0] / (float)D + 1e-5f);

    for (int i = tid; i < D; i += 256) {
        float y = (xr[i] - mean) * rstd * gamma[i] + beta[i];
        if (outF) outF[row * D + i] = y;
        if (outB) outB[row * D + i] = (__bf16)y;
    }
}

// ---------------------------------------------------------------------------
// Elementwise helpers
// ---------------------------------------------------------------------------
__global__ __launch_bounds__(256) void cvt_bf16_kernel(
    const float* __restrict__ src, __bf16* __restrict__ dst, long n)
{
    long i = (long)blockIdx.x * 256 + threadIdx.x;
    if (i < n) dst[i] = (__bf16)src[i];
}

// src [R,C] row-major fp32 -> dst [C,R] row-major bf16
__global__ __launch_bounds__(256) void cvt_bf16_T_kernel(
    const float* __restrict__ src, __bf16* __restrict__ dst, int R, int C)
{
    long i = (long)blockIdx.x * 256 + threadIdx.x;
    if (i < (long)R * C) {
        int r = (int)(i / C), c = (int)(i % C);
        dst[(long)c * R + r] = (__bf16)src[i];
    }
}

// Normalized im2col: images[B,3,224,224] -> Xp bf16 [PM, 768]
__global__ __launch_bounds__(256) void im2col_kernel(
    const float* __restrict__ img, __bf16* __restrict__ xp)
{
    long i = (long)blockIdx.x * 256 + threadIdx.x;
    if (i >= (long)PM * DD) return;
    int col = (int)(i % DD); long row = i / DD;
    float val = 0.f;
    if (row < PV) {
        int b = (int)(row / 196), p = (int)(row % 196);
        int py = p / 14, px = p % 14;
        int c = col >> 8, rem = col & 255, ph = rem >> 4, pw = rem & 15;
        const float MEANC[3] = {0.48145466f, 0.4578275f, 0.40821073f};
        const float STDC[3]  = {0.26862954f, 0.26130258f, 0.27577711f};
        float x = img[(((long)b * 3 + c) * 224 + (py * 16 + ph)) * 224 +
                      (px * 16 + pw)];
        val = (x - MEANC[c]) / STDC[c];
    }
    xp[i] = (__bf16)val;
}

// X[b*SP+s, :] = (s==0 ? cls : patch[b,s-1]) + pos[s]; padded rows -> 0
__global__ __launch_bounds__(256) void assemble_kernel(
    const float* __restrict__ patch, const float* __restrict__ cls,
    const float* __restrict__ pos, float* __restrict__ X)
{
    long i = (long)blockIdx.x * 256 + threadIdx.x;
    if (i >= (long)MP * DD) return;
    int d = (int)(i % DD); long r = i / DD;
    int s = (int)(r % SP);
    int b = (int)(r / SP);
    float v = 0.f;
    if (s == 0)       v = cls[d] + pos[d];
    else if (s < SS)  v = patch[((long)b * 196 + (s - 1)) * DD + d] +
                          pos[(long)s * DD + d];
    X[i] = v;
}

// qkv [MP,2304] -> Q,K bf16 [BH,SP,64] (Q scaled 1/8), Vt bf16 [BH,64,SP]
__global__ __launch_bounds__(256) void qkv_rearrange_kernel(
    const float* __restrict__ qkv, __bf16* __restrict__ Q,
    __bf16* __restrict__ Kb, __bf16* __restrict__ Vt)
{
    long i = (long)blockIdx.x * 256 + threadIdx.x;
    if (i >= (long)BH * SP * HDIM) return;
    int d = (int)(i % HDIM);
    long t = i / HDIM;
    int s = (int)(t % SP);
    int bh = (int)(t / SP);
    int b = bh / HH, h = bh % HH;
    float q = 0.f, k = 0.f, v = 0.f;
    if (s < SS) {
        long ro = ((long)b * SP + s) * TD;
        int col = h * HDIM + d;
        q = qkv[ro + col] * 0.125f;        // HD^-0.5
        k = qkv[ro + DD + col];
        v = qkv[ro + 2 * DD + col];
    }
    Q[i]  = (__bf16)q;
    Kb[i] = (__bf16)k;
    Vt[((long)bh * HDIM + d) * SP + s] = (__bf16)v;
}

// Row softmax over SS valid cols of SP; masked/padded -> 0; bf16 out.
__global__ __launch_bounds__(256) void softmax_kernel(
    const float* __restrict__ scores, __bf16* __restrict__ attn)
{
    const long row = blockIdx.x;               // bh*SP + q
    const int q = (int)(row % SP);
    const float* sr = scores + row * SP;
    __bf16* ar = attn + row * SP;
    const int tid = threadIdx.x;

    if (q >= SS) {                              // padded query row (uniform)
        if (tid < SP) ar[tid] = (__bf16)0.f;
        return;
    }
    __shared__ float red[256];
    float v = (tid < SS) ? sr[tid] : -1e30f;
    red[tid] = v; __syncthreads();
    for (int o = 128; o > 0; o >>= 1) {
        if (tid < o) red[tid] = fmaxf(red[tid], red[tid + o]);
        __syncthreads();
    }
    const float mx = red[0]; __syncthreads();
    float e = (tid < SS) ? __expf(v - mx) : 0.f;
    red[tid] = e; __syncthreads();
    for (int o = 128; o > 0; o >>= 1) {
        if (tid < o) red[tid] += red[tid + o];
        __syncthreads();
    }
    const float inv = 1.f / red[0];
    if (tid < SP) ar[tid] = (__bf16)(e * inv);
}

// obuf [BH,SP,64] fp32 -> attn_cat bf16 [MP,768] (concat heads)
__global__ __launch_bounds__(256) void merge_heads_kernel(
    const float* __restrict__ obuf, __bf16* __restrict__ cat)
{
    long i = (long)blockIdx.x * 256 + threadIdx.x;
    if (i >= (long)MP * DD) return;
    int col = (int)(i % DD); long r = i / DD;
    int s = (int)(r % SP), b = (int)(r / SP);
    int h = col / HDIM, d = col % HDIM;
    cat[i] = (__bf16)obuf[(((long)(b * HH + h)) * SP + s) * HDIM + d];
}

// Gather cls tokens (row b*SP) into [64,768] fp32, rows 32..63 zero.
__global__ __launch_bounds__(256) void gather_cls_kernel(
    const float* __restrict__ X, float* __restrict__ cls)
{
    int i = blockIdx.x * 256 + threadIdx.x;
    if (i >= 64 * DD) return;
    int r = i / DD, d = i % DD;
    cls[i] = (r < BB) ? X[((long)r * SP) * DD + d] : 0.f;
}

// ---------------------------------------------------------------------------
// Host driver
// ---------------------------------------------------------------------------
extern "C" void kernel_launch(void* const* d_in, const int* in_sizes, int n_in,
                              void* d_out, int out_size, void* d_ws, size_t ws_size,
                              hipStream_t stream)
{
    (void)in_sizes; (void)n_in; (void)out_size; (void)ws_size;

    const float* images   = (const float*)d_in[0];
    const float* conv_w   = (const float*)d_in[1];
    const float* cls_emb  = (const float*)d_in[2];
    const float* pos_emb  = (const float*)d_in[3];
    const float* ln_pre_s = (const float*)d_in[4];
    const float* ln_pre_b = (const float*)d_in[5];
    const float* ln1_s    = (const float*)d_in[6];
    const float* ln1_b    = (const float*)d_in[7];
    const float* qkv_w    = (const float*)d_in[8];
    const float* qkv_b    = (const float*)d_in[9];
    const float* out_w    = (const float*)d_in[10];
    const float* out_b    = (const float*)d_in[11];
    const float* ln2_s    = (const float*)d_in[12];
    const float* ln2_b    = (const float*)d_in[13];
    const float* fc_w     = (const float*)d_in[14];
    const float* fc_b     = (const float*)d_in[15];
    const float* proj_w   = (const float*)d_in[16];
    const float* proj_b   = (const float*)d_in[17];
    const float* ln_post_s= (const float*)d_in[18];
    const float* ln_post_b= (const float*)d_in[19];
    const float* vis_proj = (const float*)d_in[20];

    char* p = (char*)d_ws;
    auto alloc = [&](size_t bytes) -> void* {
        void* r = (void*)p;
        p += (bytes + 255) & ~(size_t)255;
        return r;
    };

    // bf16 weight copies
    __bf16* w_conv = (__bf16*)alloc((size_t)DD * DD * 2);
    __bf16* w_qkv  = (__bf16*)alloc((size_t)LL * TD * DD * 2);
    __bf16* w_out  = (__bf16*)alloc((size_t)LL * DD * DD * 2);
    __bf16* w_fc   = (__bf16*)alloc((size_t)LL * FF * DD * 2);
    __bf16* w_prj  = (__bf16*)alloc((size_t)LL * DD * FF * 2);
    __bf16* w_vp   = (__bf16*)alloc((size_t)OO * DD * 2);
    // activations / scratch
    float*  X      = (float*) alloc((size_t)MP * DD * 4);
    __bf16* lnbuf  = (__bf16*)alloc((size_t)MP * DD * 2);
    float*  qkvbuf = (float*) alloc((size_t)MP * TD * 4);
    __bf16* Qb     = (__bf16*)alloc((size_t)BH * SP * HDIM * 2);
    __bf16* Kb     = (__bf16*)alloc((size_t)BH * SP * HDIM * 2);
    __bf16* Vt     = (__bf16*)alloc((size_t)BH * HDIM * SP * 2);
    float*  scores = (float*) alloc((size_t)BH * SP * SP * 4);
    __bf16* attn   = (__bf16*)alloc((size_t)BH * SP * SP * 2);
    float*  obuf   = (float*) alloc((size_t)BH * SP * HDIM * 4);
    __bf16* acat   = (__bf16*)alloc((size_t)MP * DD * 2);
    __bf16* hidden = (__bf16*)alloc((size_t)MP * FF * 2);
    __bf16* xp     = (__bf16*)alloc((size_t)PM * DD * 2);
    float*  pout   = (float*) alloc((size_t)PM * DD * 4);
    float*  clsf   = (float*) alloc((size_t)64 * DD * 4);
    __bf16* clsb   = (__bf16*)alloc((size_t)64 * DD * 2);

    auto cdiv = [](long a, long b) -> unsigned { return (unsigned)((a + b - 1) / b); };

    auto gemm = [&](const __bf16* A, long sA, const __bf16* Bw, long sB,
                    float* oF, __bf16* oB, long sC, const float* bias,
                    const float* res, long sR, int M, int N, int K,
                    int Mv, int act, int nb) {
        dim3 g((unsigned)((N + 127) / 128), (unsigned)((M + 63) / 64), (unsigned)nb);
        gemm_bf16_kernel<<<g, 256, 0, stream>>>(A, sA, Bw, sB, oF, oB, sC,
                                                bias, res, sR, M, N, K, Mv, act);
    };

    // ---- weight conversion (fp32 -> bf16, row-major [N,K]) ----
    cvt_bf16_kernel<<<cdiv((long)DD * DD, 256), 256, 0, stream>>>(conv_w, w_conv, (long)DD * DD);
    cvt_bf16_kernel<<<cdiv((long)LL * TD * DD, 256), 256, 0, stream>>>(qkv_w, w_qkv, (long)LL * TD * DD);
    cvt_bf16_kernel<<<cdiv((long)LL * DD * DD, 256), 256, 0, stream>>>(out_w, w_out, (long)LL * DD * DD);
    cvt_bf16_kernel<<<cdiv((long)LL * FF * DD, 256), 256, 0, stream>>>(fc_w, w_fc, (long)LL * FF * DD);
    cvt_bf16_kernel<<<cdiv((long)LL * DD * FF, 256), 256, 0, stream>>>(proj_w, w_prj, (long)LL * DD * FF);
    cvt_bf16_T_kernel<<<cdiv((long)DD * OO, 256), 256, 0, stream>>>(vis_proj, w_vp, DD, OO);

    // ---- patch embedding: im2col + GEMM [6336 x 768 x 768] ----
    im2col_kernel<<<cdiv((long)PM * DD, 256), 256, 0, stream>>>(images, xp);
    gemm(xp, 0, w_conv, 0, pout, nullptr, 0, nullptr, nullptr, 0,
         PM, DD, DD, PV, 0, 1);
    assemble_kernel<<<cdiv((long)MP * DD, 256), 256, 0, stream>>>(pout, cls_emb, pos_emb, X);
    // ln_pre (in-place on residual stream)
    layernorm_kernel<<<MP, 256, 0, stream>>>(X, ln_pre_s, ln_pre_b, X, nullptr, DD);

    // ---- transformer blocks ----
    for (int l = 0; l < LL; ++l) {
        layernorm_kernel<<<MP, 256, 0, stream>>>(X, ln1_s + (long)l * DD,
                                                 ln1_b + (long)l * DD,
                                                 nullptr, lnbuf, DD);
        gemm(lnbuf, 0, w_qkv + (long)l * TD * DD, 0, qkvbuf, nullptr, 0,
             qkv_b + (long)l * TD, nullptr, 0, MP, TD, DD, MP, 0, 1);
        qkv_rearrange_kernel<<<cdiv((long)BH * SP * HDIM, 256), 256, 0, stream>>>(
            qkvbuf, Qb, Kb, Vt);
        // scores = Q @ K^T (batched over 384 heads)
        gemm(Qb, (long)SP * HDIM, Kb, (long)SP * HDIM, scores, nullptr,
             (long)SP * SP, nullptr, nullptr, 0, SP, SP, HDIM, SP, 0, BH);
        softmax_kernel<<<BH * SP, 256, 0, stream>>>(scores, attn);
        // O = attn @ V  (B-operand is V^T so fragments are contiguous)
        gemm(attn, (long)SP * SP, Vt, (long)HDIM * SP, obuf, nullptr,
             (long)SP * HDIM, nullptr, nullptr, 0, SP, HDIM, SP, SP, 0, BH);
        merge_heads_kernel<<<cdiv((long)MP * DD, 256), 256, 0, stream>>>(obuf, acat);
        // out-proj + residual (accumulate into X)
        gemm(acat, 0, w_out + (long)l * DD * DD, 0, X, nullptr, 0,
             out_b + (long)l * DD, X, 0, MP, DD, DD, MP, 0, 1);
        // MLP
        layernorm_kernel<<<MP, 256, 0, stream>>>(X, ln2_s + (long)l * DD,
                                                 ln2_b + (long)l * DD,
                                                 nullptr, lnbuf, DD);
        gemm(lnbuf, 0, w_fc + (long)l * FF * DD, 0, nullptr, hidden, 0,
             fc_b + (long)l * FF, nullptr, 0, MP, FF, DD, MP, 1, 1);
        gemm(hidden, 0, w_prj + (long)l * DD * FF, 0, X, nullptr, 0,
             proj_b + (long)l * DD, X, 0, MP, DD, FF, MP, 0, 1);
    }

    // ---- head: ln_post on cls token, project to 512 ----
    gather_cls_kernel<<<cdiv(64 * DD, 256), 256, 0, stream>>>(X, clsf);
    layernorm_kernel<<<64, 256, 0, stream>>>(clsf, ln_post_s, ln_post_b,
                                             nullptr, clsb, DD);
    gemm(clsb, 0, w_vp, 0, (float*)d_out, nullptr, 0, nullptr, nullptr, 0,
         64, OO, DD, BB, 0, 1);
}
